// SpaceFillingVQ_62139586838843
// MI455X (gfx1250) — matrix operand order repaired and checked by reading, not compiled, not run on hardware
//
#include <hip/hip_runtime.h>
#include <math.h>

// Problem constants (fixed by the reference).
#define NROWS 65536
#define DDIM  64
#define EPAD  4096     // padded entry count (power of two, multiple of 16)
#define EREAL 4095     // actual dithered entries
#define ETILES (EPAD / 16)
#define EALLOC (EPAD + 16)   // +1 tile so the software pipeline may over-read

typedef __attribute__((ext_vector_type(8)))  float  v8f;
typedef __attribute__((ext_vector_type(16))) __bf16 v16bf;
typedef __attribute__((ext_vector_type(8)))  __bf16 v8bf;

// ---------------------------------------------------------------------------
// WMMA wrapper: D = A(16x32 bf16) * B(32x16 bf16) + C(16x16 f32)
// ---------------------------------------------------------------------------
static __device__ __forceinline__ v8f wmma_bf16(v16bf a, v16bf b, v8f c) {
    return __builtin_amdgcn_wmma_f32_16x16x32_bf16(
        /*neg_a=*/false, a, /*neg_b=*/false, b,
        /*c_mod=*/(short)0, c, /*reuse_a=*/false, /*reuse_b=*/false);
}

// A fragment (16x32, row-major source, ld elements per row), per ISA 7.12.2:
// lanes 0-15: M=lane,    elems 0..7 = K 0..7,  elems 8..15 = K 16..23
// lanes16-31: M=lane-16, elems 0..7 = K 8..15, elems 8..15 = K 24..31
static __device__ __forceinline__ v16bf load_a_frag(const __bf16* __restrict__ base,
                                                    int ld, int lane, int k0) {
    const int m    = lane & 15;
    const int koff = (lane >> 4) << 3;          // 0 or 8
    const __bf16* p = base + (size_t)m * ld + k0 + koff;
    v8bf lo = *reinterpret_cast<const v8bf*>(p);
    v8bf hi = *reinterpret_cast<const v8bf*>(p + 16);
    v16bf r;
#pragma unroll
    for (int i = 0; i < 8; ++i) { r[i] = lo[i]; r[8 + i] = hi[i]; }
    return r;
}

// B fragment (32x16). B[k][n] = codebook[n0+n][k0+k] (codebook row-major, ld).
// lanes 0-15: N=lane, elems 0..7 = K 0..7, 8..15 = K 16..23 ; lanes 16-31: N=lane-16, elems 0..7 = K 8..15, 8..15 = K 24..31
static __device__ __forceinline__ v16bf load_b_frag(const __bf16* __restrict__ base,
                                                    int ld, int lane, int k0) {
    const int n    = lane & 15;
    const int koff = (lane >> 4) << 3;          // 0 or 8
    const __bf16* p = base + (size_t)n * ld + k0 + koff;
    v8bf lo = *reinterpret_cast<const v8bf*>(p);
    v8bf hi = *reinterpret_cast<const v8bf*>(p + 16);
    v16bf r;
#pragma unroll
    for (int i = 0; i < 8; ++i) { r[i] = lo[i]; r[8 + i] = hi[i]; }
    return r;
}

// Load the 4 B fragments for one 16-entry tile: {hi k0, hi k1, lo k0, lo k1}.
static __device__ __forceinline__ void load_b_set(const __bf16* __restrict__ Chi,
                                                  const __bf16* __restrict__ Clo,
                                                  int n0, int lane, v16bf (&B)[4]) {
    const __bf16* bh = Chi + (size_t)n0 * DDIM;
    const __bf16* bl = Clo + (size_t)n0 * DDIM;
    B[0] = load_b_frag(bh, DDIM, lane, 0);
    B[1] = load_b_frag(bh, DDIM, lane, 32);
    B[2] = load_b_frag(bl, DDIM, lane, 0);
    B[3] = load_b_frag(bl, DDIM, lane, 32);
}

// 12 WMMAs: two 16-row groups against one 16-entry B tile, then argmax update.
static __device__ __forceinline__ void compute_tile(
    const v16bf (&ah)[2][2], const v16bf (&al)[2][2], const v16bf (&B)[4],
    int n0, int col_in_tile, const float* __restrict__ halfc2,
    float (&best_val)[2][8], int (&best_idx)[2][8]) {

    const float hsq = halfc2[n0 + col_in_tile];
    const int col = n0 + col_in_tile;
#pragma unroll
    for (int g = 0; g < 2; ++g) {
        v8f acc = { -hsq, -hsq, -hsq, -hsq, -hsq, -hsq, -hsq, -hsq };
        acc = wmma_bf16(ah[g][0], B[0], acc);   // hi*hi  (K 0..31)
        acc = wmma_bf16(ah[g][0], B[2], acc);   // hi*lo
        acc = wmma_bf16(al[g][0], B[0], acc);   // lo*hi
        acc = wmma_bf16(ah[g][1], B[1], acc);   // hi*hi  (K 32..63)
        acc = wmma_bf16(ah[g][1], B[3], acc);   // hi*lo
        acc = wmma_bf16(al[g][1], B[1], acc);   // lo*hi
#pragma unroll
        for (int r = 0; r < 8; ++r) {
            const float s = acc[r];
            if (s > best_val[g][r]) { best_val[g][r] = s; best_idx[g][r] = col; }
        }
    }
}

// ---------------------------------------------------------------------------
// Kernel 1: build dithered codebook rows, bf16 hi/lo split, 0.5*||c||^2,
//           keep an fp32 copy for the gather, zero the histogram.
// grid = EALLOC blocks, 64 threads.
// ---------------------------------------------------------------------------
__global__ void vq_prep_codebook(const float* __restrict__ cb,
                                 const float* __restrict__ dither,
                                 __bf16* __restrict__ Chi,
                                 __bf16* __restrict__ Clo,
                                 float*  __restrict__ Cf32,
                                 float*  __restrict__ halfc2,
                                 int*    __restrict__ counts) {
    const int e = blockIdx.x;
    const int t = threadIdx.x;
    __shared__ float red[64];

    float v = 0.0f;
    if (e < EREAL) {
        const float r = dither[e];
        v = (1.0f - r) * cb[e * DDIM + t] + r * cb[(e + 1) * DDIM + t];
    }
    if (e < EPAD) Cf32[(size_t)e * DDIM + t] = v;
    const __bf16 h = (__bf16)v;
    Chi[(size_t)e * DDIM + t] = h;
    Clo[(size_t)e * DDIM + t] = (__bf16)(v - (float)h);

    red[t] = v * v;
    __syncthreads();
#pragma unroll
    for (int s = 32; s > 0; s >>= 1) {
        if (t < s) red[t] += red[t + s];
        __syncthreads();
    }
    if (t == 0) {
        halfc2[e] = (e < EREAL) ? 0.5f * red[0] : 1e30f;  // pad entries never win
        if (e < EPAD) counts[e] = 0;
    }
}

// ---------------------------------------------------------------------------
// Kernel 2: split fp32 inputs into bf16 hi + residual lo.
// ---------------------------------------------------------------------------
__global__ void vq_split_x(const float* __restrict__ X,
                           __bf16* __restrict__ Xhi,
                           __bf16* __restrict__ Xlo, int n) {
    const int i = blockIdx.x * blockDim.x + threadIdx.x;
    if (i < n) {
        const float f = X[i];
        const __bf16 h = (__bf16)f;
        Xhi[i] = h;
        Xlo[i] = (__bf16)(f - (float)h);
    }
}

// ---------------------------------------------------------------------------
// Kernel 3: fused GEMM + argmax.
// Wave = 32 rows (two 16-row groups), block = 8 waves = 256 rows.
// score(i,e) = x_i . c_e - 0.5*||c_e||^2 via split-bf16 WMMA:
//   xh*ch + xh*cl + xl*ch per 32-wide K chunk -> 12 WMMAs per B tile.
// B tiles are double-buffered (2x unrolled loop) to hide L2 latency.
// ---------------------------------------------------------------------------
__global__ __launch_bounds__(256) void vq_gemm_argmax(
    const __bf16* __restrict__ Xhi, const __bf16* __restrict__ Xlo,
    const __bf16* __restrict__ Chi, const __bf16* __restrict__ Clo,
    const float*  __restrict__ halfc2,
    int* __restrict__ idx_out, int* __restrict__ counts) {

    const int lane = threadIdx.x & 31;
    const int wave = threadIdx.x >> 5;
    const int row0 = blockIdx.x * 256 + wave * 32;
    const int col_in_tile = lane & 15;

    // A fragments: [row-group][k-chunk], hi and lo parts. Register-resident.
    v16bf ah[2][2], al[2][2];
#pragma unroll
    for (int g = 0; g < 2; ++g) {
        const __bf16* Ah = Xhi + (size_t)(row0 + g * 16) * DDIM;
        const __bf16* Al = Xlo + (size_t)(row0 + g * 16) * DDIM;
        ah[g][0] = load_a_frag(Ah, DDIM, lane, 0);
        ah[g][1] = load_a_frag(Ah, DDIM, lane, 32);
        al[g][0] = load_a_frag(Al, DDIM, lane, 0);
        al[g][1] = load_a_frag(Al, DDIM, lane, 32);
    }

    float best_val[2][8];
    int   best_idx[2][8];
#pragma unroll
    for (int g = 0; g < 2; ++g)
#pragma unroll
        for (int r = 0; r < 8; ++r) { best_val[g][r] = -3.0e38f; best_idx[g][r] = 0; }

    // Software pipeline: Ba holds tile et, Bb holds tile et+1.
    v16bf Ba[4], Bb[4];
    load_b_set(Chi, Clo, 0, lane, Ba);

    for (int et = 0; et < ETILES; et += 2) {
        load_b_set(Chi, Clo, (et + 1) * 16, lane, Bb);
        compute_tile(ah, al, Ba, et * 16, col_in_tile, halfc2, best_val, best_idx);
        load_b_set(Chi, Clo, (et + 2) * 16, lane, Ba);   // over-read lands in pad tile
        compute_tile(ah, al, Bb, (et + 1) * 16, col_in_tile, halfc2, best_val, best_idx);
    }

    // Reduce across the 16 lanes of each half-wave (columns of the tile).
    // xor masks 1..8 keep bit4 fixed, so halves reduce independently.
#pragma unroll
    for (int off = 8; off > 0; off >>= 1) {
#pragma unroll
        for (int g = 0; g < 2; ++g)
#pragma unroll
            for (int r = 0; r < 8; ++r) {
                const float ov = __shfl_xor(best_val[g][r], off, 32);
                const int   oi = __shfl_xor(best_idx[g][r], off, 32);
                if (ov > best_val[g][r] ||
                    (ov == best_val[g][r] && oi < best_idx[g][r])) {
                    best_val[g][r] = ov; best_idx[g][r] = oi;
                }
            }
    }

    if (col_in_tile == 0) {
        const int half = lane >> 4;                 // 0: rows 0..7, 1: rows 8..15
#pragma unroll
        for (int g = 0; g < 2; ++g)
#pragma unroll
            for (int r = 0; r < 8; ++r) {
                const int row = row0 + g * 16 + half * 8 + r;
                const int e   = best_idx[g][r];
                idx_out[row] = e;
                atomicAdd(&counts[e], 1);
            }
    }
}

// ---------------------------------------------------------------------------
// Kernel 4: gather quantized rows + emit indices (as float, per output dtype).
// grid = NROWS blocks, 64 threads.
// ---------------------------------------------------------------------------
__global__ void vq_gather(const int* __restrict__ idx,
                          const float* __restrict__ Cf32,
                          float* __restrict__ quant,
                          float* __restrict__ idx_f) {
    const int i = blockIdx.x;
    const int t = threadIdx.x;
    const int e = idx[i];
    quant[(size_t)i * DDIM + t] = Cf32[(size_t)e * DDIM + t];
    if (t == 0) idx_f[i] = (float)e;
}

// ---------------------------------------------------------------------------
// Kernel 5: perplexity = exp(-sum p*log(p+1e-10)).
// ---------------------------------------------------------------------------
__global__ void vq_perplexity(const int* __restrict__ counts,
                              float* __restrict__ out) {
    __shared__ float red[256];
    const int t = threadIdx.x;
    float s = 0.0f;
    for (int e = t; e < EPAD; e += 256) {
        const float p = (float)counts[e] * (1.0f / (float)NROWS);
        s += p * __logf(p + 1e-10f);
    }
    red[t] = s;
    __syncthreads();
#pragma unroll
    for (int k = 128; k > 0; k >>= 1) {
        if (t < k) red[t] += red[t + k];
        __syncthreads();
    }
    if (t == 0) out[0] = __expf(-red[0]);
}

// ---------------------------------------------------------------------------
extern "C" void kernel_launch(void* const* d_in, const int* in_sizes, int n_in,
                              void* d_out, int out_size, void* d_ws, size_t ws_size,
                              hipStream_t stream) {
    const float* X      = (const float*)d_in[0];   // NROWS x DDIM
    const float* CB     = (const float*)d_in[1];   // EPAD x DDIM
    const float* DITHER = (const float*)d_in[2];   // EREAL

    // Workspace carve-up (256B-aligned regions), ~18.4 MB total.
    char* ws = (char*)d_ws;
    size_t off = 0;
    auto take = [&](size_t bytes) -> char* {
        char* p = ws + off;
        off += (bytes + 255) & ~(size_t)255;
        return p;
    };
    __bf16* Xhi    = (__bf16*)take((size_t)NROWS  * DDIM * sizeof(__bf16));
    __bf16* Xlo    = (__bf16*)take((size_t)NROWS  * DDIM * sizeof(__bf16));
    __bf16* Chi    = (__bf16*)take((size_t)EALLOC * DDIM * sizeof(__bf16));
    __bf16* Clo    = (__bf16*)take((size_t)EALLOC * DDIM * sizeof(__bf16));
    float*  Cf32   = (float*) take((size_t)EPAD   * DDIM * sizeof(float));
    float*  halfc2 = (float*) take((size_t)EALLOC * sizeof(float));
    int*    counts = (int*)   take((size_t)EPAD * sizeof(int));
    int*    idx    = (int*)   take((size_t)NROWS * sizeof(int));
    (void)ws_size; (void)in_sizes; (void)n_in; (void)out_size;

    float* out_quant = (float*)d_out;                       // NROWS*DDIM
    float* out_perp  = out_quant + (size_t)NROWS * DDIM;    // 1
    float* out_idx   = out_perp + 1;                        // NROWS

    vq_prep_codebook<<<EALLOC, 64, 0, stream>>>(CB, DITHER, Chi, Clo, Cf32, halfc2, counts);

    const int nelem = NROWS * DDIM;
    vq_split_x<<<(nelem + 255) / 256, 256, 0, stream>>>(X, Xhi, Xlo, nelem);

    vq_gemm_argmax<<<NROWS / 256, 256, 0, stream>>>(Xhi, Xlo, Chi, Clo, halfc2, idx, counts);

    vq_gather<<<NROWS, 64, 0, stream>>>(idx, Cf32, out_quant, out_idx);

    vq_perplexity<<<1, 256, 0, stream>>>(counts, out_perp);
}
